// Attention_1726576857899
// MI455X (gfx1250) — matrix operand order repaired and checked by reading, not compiled, not run on hardware
//
#include <hip/hip_runtime.h>
#include <hip/hip_bf16.h>

typedef __bf16 bf16;
typedef __attribute__((ext_vector_type(16))) __bf16 v16bf;
typedef __attribute__((ext_vector_type(8)))  __bf16 v8bf;
typedef __attribute__((ext_vector_type(8)))  float  v8f;

#define DIM   512
#define SEQ   2048
#define BATCH 4
#define HEADS 8
#define DHEAD 64
#define ROWS  (BATCH*SEQ)   /* 8192 */
#define NQKV  (3*DIM)       /* 1536 */
#define ATT_SCALE 0.125f    /* 64^-0.5 */
#define KTILE 64
#define NTILES (SEQ / KTILE)   /* 32 */

// ---- fragment helpers -------------------------------------------------------
// v16bf fragment = two contiguous 16-byte halves (per-lane), per CDNA5 ISA
// 16-bit A/B VGPR layouts.
static __device__ __forceinline__ v16bf load_frag16(const bf16* p0, const bf16* p1) {
  v8bf lo = *(const v8bf*)p0;
  v8bf hi = *(const v8bf*)p1;
  v16bf r;
#pragma unroll
  for (int i = 0; i < 8; ++i) { r[i] = lo[i]; r[i + 8] = hi[i]; }
  return r;
}

static __device__ __forceinline__ v8f wmma_bf16(v16bf a, v16bf b, v8f c) {
  // 8 args: (neg_a, A, neg_b, B, c_mod, C, reuse_a, reuse_b)
  return __builtin_amdgcn_wmma_f32_16x16x32_bf16(false, a, false, b, (short)0, c,
                                                 false, false);
}

// CDNA5 async copy: 16 bytes global -> LDS per lane, tracked by ASYNCcnt.
// LDS operand = low 32 bits of the generic pointer (= addrspace(3) byte offset).
static __device__ __forceinline__ void async_copy_b128(void* lds_dst, const void* gsrc) {
  unsigned lds_off = (unsigned)(size_t)lds_dst;
  unsigned long long ga = (unsigned long long)(size_t)gsrc;
  asm volatile("global_load_async_to_lds_b128 %0, %1, off"
               :: "v"(lds_off), "v"(ga) : "memory");
}
static __device__ __forceinline__ void async_wait0() {
  asm volatile("s_wait_asynccnt 0x0" ::: "memory");
}

// ---- kernel 1: fp32 [K,N] -> bf16 [N,K] (weight cast + transpose) ----------
// K is always 512 here (inner dim of both weight matrices).
__global__ __launch_bounds__(256) void transpose_bf16_kernel(
    const float* __restrict__ in, bf16* __restrict__ out, int N) {
  int idx = blockIdx.x * 256 + threadIdx.x;   // grid sized exactly N*512/256
  int n = idx >> 9;
  int k = idx & 511;
  out[(size_t)n * DIM + k] = (bf16)in[(size_t)k * N + n];
}

// ---- kernel 2: LayerNorm -> bf16 -------------------------------------------
__global__ __launch_bounds__(256) void ln_kernel(
    const float* __restrict__ x, const float* __restrict__ gamma,
    const float* __restrict__ beta, bf16* __restrict__ xn) {
  __shared__ float red[16];
  const int row = blockIdx.x;
  const int t = threadIdx.x;
  const float* xr = x + (size_t)row * DIM;
  float a0 = xr[t], a1 = xr[t + 256];
  float s = a0 + a1, s2 = a0 * a0 + a1 * a1;
#pragma unroll
  for (int off = 16; off >= 1; off >>= 1) {
    s  += __shfl_xor(s,  off, 32);
    s2 += __shfl_xor(s2, off, 32);
  }
  const int wave = t >> 5;
  if ((t & 31) == 0) { red[wave] = s; red[8 + wave] = s2; }
  __syncthreads();
  float ts = 0.f, ts2 = 0.f;
#pragma unroll
  for (int i = 0; i < 8; ++i) { ts += red[i]; ts2 += red[8 + i]; }
  const float mu  = ts  * (1.0f / DIM);
  const float var = ts2 * (1.0f / DIM) - mu * mu;
  const float r = rsqrtf(var + 1e-5f);
  bf16* o = xn + (size_t)row * DIM;
  o[t]       = (bf16)((a0 - mu) * r * gamma[t]       + beta[t]);
  o[t + 256] = (bf16)((a1 - mu) * r * gamma[t + 256] + beta[t + 256]);
}

// ---- kernel 3: QKV GEMM (bf16 WMMA), scatter into Q/K row-major + V^T ------
__global__ __launch_bounds__(256) void qkv_gemm_kernel(
    const bf16* __restrict__ Xn,    // [ROWS, DIM]
    const bf16* __restrict__ WT,    // [NQKV, DIM]  (Wqkv transposed)
    bf16* __restrict__ Qb,          // [B*H, SEQ, DHEAD]
    bf16* __restrict__ Kb,          // [B*H, SEQ, DHEAD]
    bf16* __restrict__ Vtb) {       // [B*H, DHEAD, SEQ]
  const int lane = threadIdx.x & 31;
  const int wave = threadIdx.x >> 5;
  const int half = lane >> 4;
  const int ln16 = lane & 15;
  const int m0 = blockIdx.y * 128 + (wave >> 1) * 32;
  const int n0 = blockIdx.x * 64  + (wave & 1)  * 32;

  v8f acc[2][2] = {};
  for (int k0 = 0; k0 < DIM; k0 += 32) {
    v16bf a[2], b[2];
#pragma unroll
    for (int r = 0; r < 2; ++r) {
      const bf16* p = Xn + (size_t)(m0 + r * 16 + ln16) * DIM + k0;
      a[r] = load_frag16(p + half * 16, p + half * 16 + 8);    // A: 16 contig K
    }
#pragma unroll
    for (int c = 0; c < 2; ++c) {
      const bf16* p = WT + (size_t)(n0 + c * 16 + ln16) * DIM + k0 + half * 16;
      b[c] = load_frag16(p, p + 8);                            // B: 16 contig K
    }
#pragma unroll
    for (int r = 0; r < 2; ++r)
#pragma unroll
      for (int c = 0; c < 2; ++c)
        acc[r][c] = wmma_bf16(a[r], b[c], acc[r][c]);
  }

  // 'which' is uniform per block: each 64-wide column block is inside Q, K or V
  const int which = (int)(blockIdx.x >> 3);     // 0=Q 1=K 2=V
  if (which == 2) {
    // V^T store: D-fragment has 8 consecutive seq rows per lane at fixed d
    // -> one contiguous 16B store per fragment per lane.
#pragma unroll
    for (int r = 0; r < 2; ++r) {
      const int b_ = (m0 + r * 16) >> 11;
      const int nseq0 = (m0 + r * 16 + half * 8) & 2047;
#pragma unroll
      for (int c = 0; c < 2; ++c) {
        const int cc = (n0 + c * 16 + ln16) & 511;
        const int h = cc >> 6, d = cc & 63;
        v8bf pk;
#pragma unroll
        for (int v = 0; v < 8; ++v) pk[v] = (bf16)acc[r][c][v];
        *(v8bf*)&Vtb[((size_t)(b_ * HEADS + h) * DHEAD + d) * SEQ + nseq0] = pk;
      }
    }
  } else {
    bf16* Dst = (which == 0) ? Qb : Kb;
#pragma unroll
    for (int r = 0; r < 2; ++r)
#pragma unroll
      for (int c = 0; c < 2; ++c) {
        const int cc = (n0 + c * 16 + ln16) & 511;
        const int h = cc >> 6, d = cc & 63;
#pragma unroll
        for (int v = 0; v < 8; ++v) {
          const int m = m0 + r * 16 + half * 8 + v;
          const int b_ = m >> 11, nseq = m & 2047;
          Dst[((size_t)(b_ * HEADS + h) * SEQ + nseq) * DHEAD + d] =
              (bf16)acc[r][c][v];
        }
      }
  }
}

// ---- kernel 4: flash attention, 64-key tiles, async double buffering -------
__global__ __launch_bounds__(256) void attn_kernel(
    const bf16* __restrict__ Qb, const bf16* __restrict__ Kb,
    const bf16* __restrict__ Vtb, bf16* __restrict__ Ob) {
  __shared__ __align__(16) bf16 sK[2][KTILE * DHEAD];  // [key][d],  2 x 8 KB
  __shared__ __align__(16) bf16 sV[2][DHEAD * KTILE];  // [d][key],  2 x 8 KB
  __shared__ __align__(16) bf16 sP[8][16 * KTILE];     // per-wave P, 16 KB

  const int tid  = threadIdx.x;
  const int lane = tid & 31, wave = tid >> 5;
  const int half = lane >> 4, ln16 = lane & 15;
  const int bh = blockIdx.y;
  const int q0 = blockIdx.x * 128 + wave * 16;

  // this wave's Q A-fragments (16 rows x 64 d = two 16x32 frags), kept in VGPRs
  const bf16* qrow = Qb + ((size_t)bh * SEQ + q0 + ln16) * DHEAD;
  v16bf qf[2];
#pragma unroll
  for (int f = 0; f < 2; ++f)
    qf[f] = load_frag16(qrow + f * 32 + half * 16, qrow + f * 32 + half * 16 + 8);

  float mi[8], li[8];
  v8f oa[4] = {};
#pragma unroll
  for (int v = 0; v < 8; ++v) { mi[v] = -1e30f; li[v] = 0.f; }

  const bf16* Kbase = Kb  + (size_t)bh * SEQ * DHEAD;
  const bf16* Vbase = Vtb + (size_t)bh * DHEAD * SEQ;

  // stage tile 0 into buffer 0 (4 async b128 per thread: 2 K + 2 V)
#pragma unroll
  for (int u = 0; u < 2; ++u) {
    const int c = tid * 2 + u;
    const int row = c >> 3, col = (c & 7) * 8;
    async_copy_b128(&sK[0][row * DHEAD + col], Kbase + (size_t)row * DHEAD + col);
    async_copy_b128(&sV[0][row * KTILE + col], Vbase + (size_t)row * SEQ + col);
  }

  for (int i = 0; i < NTILES; ++i) {
    const int cur = i & 1;
    const bf16* sKc = sK[cur];
    const bf16* sVc = sV[cur];

    async_wait0();     // this wave's copies of tile i are resident
    __syncthreads();   // everyone's copies resident; previous iteration done

    if (i + 1 < NTILES) {   // prefetch tile i+1 into the other buffer
      const int j = (i + 1) * KTILE;
#pragma unroll
      for (int u = 0; u < 2; ++u) {
        const int c = tid * 2 + u;
        const int row = c >> 3, col = (c & 7) * 8;
        async_copy_b128(&sK[cur ^ 1][row * DHEAD + col],
                        Kbase + (size_t)(j + row) * DHEAD + col);
        async_copy_b128(&sV[cur ^ 1][row * KTILE + col],
                        Vbase + (size_t)row * SEQ + j + col);
      }
    }

    // S = Q*K^T for 64 keys: 4 sub-blocks x 2 WMMA over d
    v8f s[4] = {};
#pragma unroll
    for (int kb = 0; kb < 4; ++kb)
#pragma unroll
      for (int f = 0; f < 2; ++f) {
        const bf16* kp = &sKc[(kb * 16 + ln16) * DHEAD + f * 32 + half * 16];
        s[kb] = wmma_bf16(qf[f], load_frag16(kp, kp + 8), s[kb]);
      }

#pragma unroll
    for (int kb = 0; kb < 4; ++kb)
#pragma unroll
      for (int v = 0; v < 8; ++v) s[kb][v] *= ATT_SCALE;

    // one online-softmax update for all 64 keys
    float tm[8];
#pragma unroll
    for (int v = 0; v < 8; ++v)
      tm[v] = fmaxf(fmaxf(s[0][v], s[1][v]), fmaxf(s[2][v], s[3][v]));
#pragma unroll
    for (int off = 8; off >= 1; off >>= 1)
#pragma unroll
      for (int v = 0; v < 8; ++v) tm[v] = fmaxf(tm[v], __shfl_xor(tm[v], off, 16));

    float alpha[8];
#pragma unroll
    for (int v = 0; v < 8; ++v) {
      const float nm = fmaxf(mi[v], tm[v]);
      alpha[v] = __expf(mi[v] - nm);
      mi[v] = nm;
    }
#pragma unroll
    for (int kb = 0; kb < 4; ++kb)
#pragma unroll
      for (int v = 0; v < 8; ++v) s[kb][v] = __expf(s[kb][v] - mi[v]);

    float rs[8];
#pragma unroll
    for (int v = 0; v < 8; ++v)
      rs[v] = (s[0][v] + s[1][v]) + (s[2][v] + s[3][v]);
#pragma unroll
    for (int off = 8; off >= 1; off >>= 1)
#pragma unroll
      for (int v = 0; v < 8; ++v) rs[v] += __shfl_xor(rs[v], off, 16);
#pragma unroll
    for (int v = 0; v < 8; ++v) li[v] = li[v] * alpha[v] + rs[v];
#pragma unroll
    for (int g = 0; g < 4; ++g)
#pragma unroll
      for (int v = 0; v < 8; ++v) oa[g][v] *= alpha[v];

    // stage P (D-layout -> row-major LDS); per-wave private buffer, and DS ops
    // from one wave execute in order, so no block barrier is needed here.
#pragma unroll
    for (int kb = 0; kb < 4; ++kb)
#pragma unroll
      for (int v = 0; v < 8; ++v)
        sP[wave][(half * 8 + v) * KTILE + kb * 16 + ln16] = (bf16)s[kb][v];
    asm volatile("" ::: "memory");   // keep compiler from reordering P store/load

    // O += P * V : 2 key chunks x 4 d-groups
#pragma unroll
    for (int t = 0; t < 2; ++t) {
      const bf16* pp = &sP[wave][ln16 * KTILE + t * 32];
      v16bf pa = load_frag16(pp + half * 16, pp + half * 16 + 8);
#pragma unroll
      for (int g = 0; g < 4; ++g) {
        const bf16* vp = &sVc[(g * 16 + ln16) * KTILE + t * 32 + half * 16];
        oa[g] = wmma_bf16(pa, load_frag16(vp, vp + 8), oa[g]);
      }
    }
  }

  // epilogue: O /= l, write to [b, n, h*64+d] bf16
  const int b_ = bh >> 3, h = bh & 7;
#pragma unroll
  for (int v = 0; v < 8; ++v) {
    const float inv = 1.0f / li[v];
    const int row = b_ * SEQ + q0 + half * 8 + v;
#pragma unroll
    for (int g = 0; g < 4; ++g) {
      const int col = h * DHEAD + g * 16 + ln16;
      Ob[(size_t)row * DIM + col] = (bf16)(oa[g][v] * inv);
    }
  }
}

// ---- kernel 5: output projection + bias (fp32 out) -------------------------
__global__ __launch_bounds__(256) void out_gemm_kernel(
    const bf16* __restrict__ Ob,    // [ROWS, DIM]
    const bf16* __restrict__ WT,    // [DIM, DIM] (Wout transposed)
    const float* __restrict__ bias,
    float* __restrict__ out) {
  const int lane = threadIdx.x & 31;
  const int wave = threadIdx.x >> 5;
  const int half = lane >> 4;
  const int ln16 = lane & 15;
  const int m0 = blockIdx.y * 128 + (wave >> 1) * 32;
  const int n0 = blockIdx.x * 64  + (wave & 1)  * 32;

  v8f acc[2][2] = {};
  for (int k0 = 0; k0 < DIM; k0 += 32) {
    v16bf a[2], b[2];
#pragma unroll
    for (int r = 0; r < 2; ++r) {
      const bf16* p = Ob + (size_t)(m0 + r * 16 + ln16) * DIM + k0;
      a[r] = load_frag16(p + half * 16, p + half * 16 + 8);
    }
#pragma unroll
    for (int c = 0; c < 2; ++c) {
      const bf16* p = WT + (size_t)(n0 + c * 16 + ln16) * DIM + k0 + half * 16;
      b[c] = load_frag16(p, p + 8);
    }
#pragma unroll
    for (int r = 0; r < 2; ++r)
#pragma unroll
      for (int c = 0; c < 2; ++c)
        acc[r][c] = wmma_bf16(a[r], b[c], acc[r][c]);
  }
#pragma unroll
  for (int r = 0; r < 2; ++r)
#pragma unroll
    for (int c = 0; c < 2; ++c) {
      const int col = n0 + c * 16 + ln16;
      const float bv = bias[col];
#pragma unroll
      for (int v = 0; v < 8; ++v) {
        const int m = m0 + r * 16 + half * 8 + v;
        out[(size_t)m * DIM + col] = acc[r][c][v] + bv;
      }
    }
}

// ---- host launcher ---------------------------------------------------------
extern "C" void kernel_launch(void* const* d_in, const int* in_sizes, int n_in,
                              void* d_out, int out_size, void* d_ws, size_t ws_size,
                              hipStream_t stream) {
  const float* x     = (const float*)d_in[0];
  const float* gamma = (const float*)d_in[1];
  const float* beta  = (const float*)d_in[2];
  const float* Wqkv  = (const float*)d_in[3];
  const float* Wout  = (const float*)d_in[4];
  const float* bout  = (const float*)d_in[5];
  float* out = (float*)d_out;

  char* ws = (char*)d_ws;
  size_t off = 0;
  auto carve = [&](size_t bytes) -> void* {
    void* p = ws + off;
    off = (off + bytes + 255) & ~(size_t)255;
    return p;
  };
  bf16* xn    = (bf16*)carve((size_t)ROWS * DIM * sizeof(bf16));   // 8 MB
  bf16* wqkvT = (bf16*)carve((size_t)NQKV * DIM * sizeof(bf16));   // 1.5 MB
  bf16* woutT = (bf16*)carve((size_t)DIM  * DIM * sizeof(bf16));   // 0.5 MB
  bf16* Qb    = (bf16*)carve((size_t)ROWS * DIM * sizeof(bf16));   // 8 MB
  bf16* Kb    = (bf16*)carve((size_t)ROWS * DIM * sizeof(bf16));   // 8 MB
  bf16* Vtb   = (bf16*)carve((size_t)ROWS * DIM * sizeof(bf16));   // 8 MB
  bf16* Ob    = (bf16*)carve((size_t)ROWS * DIM * sizeof(bf16));   // 8 MB
  (void)ws_size; (void)in_sizes; (void)n_in; (void)out_size;

  transpose_bf16_kernel<<<(NQKV * DIM) / 256, 256, 0, stream>>>(Wqkv, wqkvT, NQKV);
  transpose_bf16_kernel<<<(DIM  * DIM) / 256, 256, 0, stream>>>(Wout, woutT, DIM);
  ln_kernel<<<ROWS, 256, 0, stream>>>(x, gamma, beta, xn);
  qkv_gemm_kernel<<<dim3(NQKV / 64, ROWS / 128), 256, 0, stream>>>(xn, wqkvT, Qb, Kb, Vtb);
  attn_kernel<<<dim3(SEQ / 128, BATCH * HEADS), 256, 0, stream>>>(Qb, Kb, Vtb, Ob);
  out_gemm_kernel<<<dim3(DIM / 64, ROWS / 128), 256, 0, stream>>>(Ob, woutT, bout, out);
}
